// Self_Attn_59940563583875
// MI455X (gfx1250) — compile-verified, matrix operand order checked by reading, not run on hardware
//
#include <hip/hip_runtime.h>

typedef __attribute__((ext_vector_type(16))) __bf16 v16bf;
typedef __attribute__((ext_vector_type(8)))  __bf16 v8bf;
typedef __attribute__((ext_vector_type(8)))  float  v8f;

#define NB   4
#define CC   256
#define CQKD 32
#define NN   4096
#define NT   (NN/16)   // 256 spatial 16-tiles
#define MTN  (NN/32)   // 128 spatial 32-tiles
#define CTN  (CC/16)   // 16 channel 16-tiles
#define KTN  (CC/32)   // 8  GEMM K steps over C=256
#define RTN  20        // 320 projection rows / 16
#define LOG2E 1.4426950408889634f

// CDNA5 16-bit operand mapping (ISA 7.12.2): K in [0,32) ->
//   h = (K>>3)&1 ; e = (K&7) + 8*(K>>4) ; lane = (M_or_N & 15) + 16*h

// ---------------------------------------------------------------------------
// Pack x (f32 [b][c][n]) into bf16 B-operand tiles xB[b][ct][nt][lane][16]
// ---------------------------------------------------------------------------
__global__ __launch_bounds__(256) void pack_x_kernel(
    const float* __restrict__ x, __bf16* __restrict__ xB)
{
  int idx = blockIdx.x * 256 + threadIdx.x;
  int n = idx & (NN - 1);
  int r = idx >> 12;
  int c = r & (CC - 1);
  int b = r >> 8;
  float v = x[((size_t)b * CC + c) * NN + n];
  int cc = c & 31, ct = c >> 5;
  int lane = (n & 15) + 16 * ((cc >> 3) & 1);
  int e    = (cc & 7) + 8 * (cc >> 4);
  xB[((((size_t)b * KTN + ct) * NT + (n >> 4)) * 32 + lane) * 16 + e] = (__bf16)v;
}

// ---------------------------------------------------------------------------
// Pack Wq|Wk|Wv (320 rows x 256) into bf16 A-operand tiles WA[rt][ct][lane][16]
// Wq rows are pre-scaled by log2(e) so softmax runs natively in exp2 domain.
// ---------------------------------------------------------------------------
__global__ __launch_bounds__(256) void pack_w_kernel(
    const float* __restrict__ Wq, const float* __restrict__ Wk,
    const float* __restrict__ Wv, __bf16* __restrict__ WA)
{
  int idx = blockIdx.x * 256 + threadIdx.x;   // 320*256 threads
  int c = idx & (CC - 1);
  int row = idx >> 8;                          // 0..319
  float v;
  if (row < 32)       v = Wq[(size_t)row * CC + c] * LOG2E;
  else if (row < 64)  v = Wk[(size_t)(row - 32) * CC + c];
  else                v = Wv[(size_t)(row - 64) * CC + c];
  int rt = row >> 4, rl = row & 15;
  int cc = c & 31, ct = c >> 5;
  int lane = rl + 16 * ((cc >> 3) & 1);
  int e    = (cc & 7) + 8 * (cc >> 4);
  WA[(((size_t)rt * KTN + ct) * 32 + lane) * 16 + e] = (__bf16)v;
}

// ---------------------------------------------------------------------------
// Projection GEMM: one wave = one 16x16 output tile, 8 accumulating WMMAs
// over K=256, then bias + scatter through LDS into Q/K/V operand layouts.
// ---------------------------------------------------------------------------
__global__ __launch_bounds__(128) void proj_gemm_kernel(
    const __bf16* __restrict__ WA, const __bf16* __restrict__ xB,
    const float* __restrict__ bq, const float* __restrict__ bk,
    const float* __restrict__ bv,
    __bf16* __restrict__ qT, __bf16* __restrict__ kB, __bf16* __restrict__ vB)
{
  __shared__ __bf16 ldsD[4][16 * 17];
  const int lane = threadIdx.x & 31;
  const int wv   = threadIdx.x >> 5;
  const int wgl  = blockIdx.x * 4 + wv;          // 0 .. NB*RTN*NT-1
  const int nt   = wgl & (NT - 1);
  const int tmp  = wgl >> 8;                     // 0..79
  const int rt   = tmp % RTN;
  const int b    = tmp / RTN;
  const int cl   = lane & 15;
  const int h    = lane >> 4;

  v8f acc = (v8f)(0.0f);
#pragma unroll
  for (int ct = 0; ct < KTN; ++ct) {
    v16bf wa = *(const v16bf*)(WA + (((size_t)rt * KTN + ct) * 32 + lane) * 16);
    v16bf xb = *(const v16bf*)(xB + ((((size_t)b * KTN + ct) * NT + nt) * 32 + lane) * 16);
    acc = __builtin_amdgcn_wmma_f32_16x16x32_bf16(false, wa, false, xb, (short)0, acc, false, false);
  }

  // bias per W-row (D rows are r + 8h); q bias also in exp2 domain
  const float* bias; int off; float bscale;
  if (rt < 2)      { bias = bq; off = rt * 16;      bscale = LOG2E; }
  else if (rt < 4) { bias = bk; off = rt * 16 - 32; bscale = 1.0f; }
  else             { bias = bv; off = rt * 16 - 64; bscale = 1.0f; }

  __bf16* ld = &ldsD[wv][0];
#pragma unroll
  for (int r = 0; r < 8; ++r) {
    float vvv = acc[r] + bias[off + r + 8 * h] * bscale;
    ld[(r + 8 * h) * 17 + cl] = (__bf16)vvv;
  }
  __builtin_amdgcn_wave_barrier();
  asm volatile("s_wait_dscnt 0" ::: "memory");

  if (rt < 2) {                 // Q -> A-operand tiles [b][nt][lane][16]
#pragma unroll
    for (int eo = 0; eo < 8; ++eo) {
      __bf16 vv = ld[(eo + 8 * h) * 17 + cl];
      qT[(((size_t)b * NT + nt) * 32 + lane) * 16 + (eo + 8 * rt)] = vv;
    }
  } else if (rt < 4) {          // K -> B-operand tiles [b][nt][lane][16]
#pragma unroll
    for (int eo = 0; eo < 8; ++eo) {
      __bf16 vv = ld[(eo + 8 * h) * 17 + cl];
      kB[(((size_t)b * NT + nt) * 32 + lane) * 16 + (eo + 8 * (rt - 2))] = vv;
    }
  } else {                      // V^T -> B-operand tiles [b][m/32][ct][lane][16]
    const int ctv  = rt - 4;
    const int mt32 = nt >> 1;
    const int ehi  = 8 * (nt & 1);
#pragma unroll
    for (int j = 0; j < 8; ++j) {
      __bf16 vv = ld[cl * 17 + 8 * h + j];    // row=c_local, col=m_local=8h+j
      vB[((((size_t)b * MTN + mt32) * CTN + ctv) * 32 + lane) * 16 + (j + ehi)] = vv;
    }
  }
  __builtin_amdgcn_wave_barrier();
}

// ds_swizzle xor-butterfly within the 16-lane half that holds one D-matrix row
template<int MASK>
__device__ __forceinline__ float swz(float v) {
  return __builtin_bit_cast(float,
      __builtin_amdgcn_ds_swizzle(__builtin_bit_cast(int, v), (MASK << 10) | 0x1f));
}

// ---------------------------------------------------------------------------
// Flash attention: one wave = 16 query rows x FULL C=256 (16 accum tiles).
// 64-key steps: 4 energy WMMAs, one online-softmax update (exp2 domain),
// P (16x64) D-layout -> A-layout via LDS, 32 accumulating PV WMMAs.
// ---------------------------------------------------------------------------
__global__ __launch_bounds__(128) void attn_kernel(
    const float* __restrict__ x, const float* __restrict__ gamma,
    const __bf16* __restrict__ qT, const __bf16* __restrict__ kB,
    const __bf16* __restrict__ vB, float* __restrict__ out)
{
  __shared__ __bf16 pbuf[4][16 * 72];            // per-wave 16x64 P tile, stride 72
  const int lane = threadIdx.x & 31;
  const int wv   = threadIdx.x >> 5;
  const int wgl  = blockIdx.x * 4 + wv;          // 0..1023
  const int nt   = wgl & (NT - 1);
  const int b    = wgl >> 8;
  const int cl   = lane & 15;
  const int h    = lane >> 4;

  v16bf qA = *(const v16bf*)(qT + (((size_t)b * NT + nt) * 32 + lane) * 16);

  float mrow[8], lrow[8];
  v8f   O[16];
#pragma unroll
  for (int r = 0; r < 8; ++r) { mrow[r] = -3.0e38f; lrow[r] = 0.0f; }
#pragma unroll
  for (int t = 0; t < 16; ++t) O[t] = (v8f)(0.0f);

  __bf16* pb = &pbuf[wv][0];
  const v8f z = (v8f)(0.0f);

  for (int mt = 0; mt < NN / 64; ++mt) {
    const __bf16* kb = kB + (((size_t)b * NT + 4 * mt) * 32 + lane) * 16;
    v16bf k0 = *(const v16bf*)(kb);
    v16bf k1 = *(const v16bf*)(kb + 512);
    v16bf k2 = *(const v16bf*)(kb + 1024);
    v16bf k3 = *(const v16bf*)(kb + 1536);

    v8f e0 = __builtin_amdgcn_wmma_f32_16x16x32_bf16(false, qA, false, k0, (short)0, z, false, false);
    v8f e1 = __builtin_amdgcn_wmma_f32_16x16x32_bf16(false, qA, false, k1, (short)0, z, false, false);
    v8f e2 = __builtin_amdgcn_wmma_f32_16x16x32_bf16(false, qA, false, k2, (short)0, z, false, false);
    v8f e3 = __builtin_amdgcn_wmma_f32_16x16x32_bf16(false, qA, false, k3, (short)0, z, false, false);

    // prefetch next iteration's K fragments (global_prefetch_b8)
    __builtin_prefetch(kb + 2048, 0, 1);

#pragma unroll
    for (int r = 0; r < 8; ++r) {
      float tmax = fmaxf(fmaxf(e0[r], e1[r]), fmaxf(e2[r], e3[r]));
      tmax = fmaxf(tmax, swz<1>(tmax));
      tmax = fmaxf(tmax, swz<2>(tmax));
      tmax = fmaxf(tmax, swz<4>(tmax));
      tmax = fmaxf(tmax, swz<8>(tmax));
      float mnew = fmaxf(mrow[r], tmax);
      float sc   = exp2f(mrow[r] - mnew);        // energy already in log2 domain
      float p0   = exp2f(e0[r] - mnew);
      float p1   = exp2f(e1[r] - mnew);
      float p2   = exp2f(e2[r] - mnew);
      float p3   = exp2f(e3[r] - mnew);
      float rs   = (p0 + p1) + (p2 + p3);
      rs += swz<1>(rs); rs += swz<2>(rs); rs += swz<4>(rs); rs += swz<8>(rs);
      lrow[r] = lrow[r] * sc + rs;
      mrow[r] = mnew;
#pragma unroll
      for (int t = 0; t < 16; ++t) O[t][r] *= sc;
      pb[(r + 8 * h) * 72 + cl]      = (__bf16)p0;
      pb[(r + 8 * h) * 72 + 16 + cl] = (__bf16)p1;
      pb[(r + 8 * h) * 72 + 32 + cl] = (__bf16)p2;
      pb[(r + 8 * h) * 72 + 48 + cl] = (__bf16)p3;
    }

    // D-layout -> A-layout transpose of P via LDS (wave-internal)
    __builtin_amdgcn_wave_barrier();
    asm volatile("s_wait_dscnt 0" ::: "memory");
    union { v16bf v; v8bf q2[2]; } pA0, pA1;
    pA0.q2[0] = *(const v8bf*)(pb + cl * 72 + 8 * h);        // K = 0..31
    pA0.q2[1] = *(const v8bf*)(pb + cl * 72 + 16 + 8 * h);
    pA1.q2[0] = *(const v8bf*)(pb + cl * 72 + 32 + 8 * h);   // K = 32..63
    pA1.q2[1] = *(const v8bf*)(pb + cl * 72 + 48 + 8 * h);
    __builtin_amdgcn_wave_barrier();

    const __bf16* vb0 = vB + (((size_t)b * MTN + 2 * mt) * CTN) * 512 + (size_t)lane * 16;
    const __bf16* vb1 = vb0 + (size_t)CTN * 512;
    __builtin_prefetch(vb1 + (size_t)CTN * 512, 0, 1);
#pragma unroll
    for (int t = 0; t < 16; ++t) {
      v16bf vt0 = *(const v16bf*)(vb0 + (size_t)t * 512);
      O[t] = __builtin_amdgcn_wmma_f32_16x16x32_bf16(false, pA0.v, false, vt0, (short)0, O[t], false, false);
    }
#pragma unroll
    for (int t = 0; t < 16; ++t) {
      v16bf vt1 = *(const v16bf*)(vb1 + (size_t)t * 512);
      O[t] = __builtin_amdgcn_wmma_f32_16x16x32_bf16(false, pA1.v, false, vt1, (short)0, O[t], false, false);
    }
  }

  const float g = gamma[0];
#pragma unroll
  for (int r = 0; r < 8; ++r) {
    float inv  = 1.0f / lrow[r];
    int   nidx = nt * 16 + r + 8 * h;
#pragma unroll
    for (int t = 0; t < 16; ++t) {
      int    c   = t * 16 + cl;
      size_t idx = ((size_t)b * CC + c) * NN + nidx;
      out[idx] = g * (O[t][r] * inv) + x[idx];
    }
  }
}

extern "C" void kernel_launch(void* const* d_in, const int* in_sizes, int n_in,
                              void* d_out, int out_size, void* d_ws, size_t ws_size,
                              hipStream_t stream) {
  const float* x     = (const float*)d_in[0];
  const float* Wq    = (const float*)d_in[1];
  const float* bq    = (const float*)d_in[2];
  const float* Wk    = (const float*)d_in[3];
  const float* bk    = (const float*)d_in[4];
  const float* Wv    = (const float*)d_in[5];
  const float* bv    = (const float*)d_in[6];
  const float* gamma = (const float*)d_in[7];
  float* out = (float*)d_out;

  __bf16* qT  = (__bf16*)d_ws;                          // 1 MB
  __bf16* kBp = qT  + (size_t)NB * NN * CQKD;           // 1 MB
  __bf16* vBp = kBp + (size_t)NB * NN * CQKD;           // 8 MB
  __bf16* xBp = vBp + (size_t)NB * NN * CC;             // 8 MB
  __bf16* WAp = xBp + (size_t)NB * NN * CC;             // 160 KB

  pack_x_kernel<<<dim3(NB * CC * NN / 256), dim3(256), 0, stream>>>(x, xBp);
  pack_w_kernel<<<dim3(RTN * 16 * CC / 256), dim3(256), 0, stream>>>(Wq, Wk, Wv, WAp);
  proj_gemm_kernel<<<dim3(NB * RTN * NT / 4), dim3(128), 0, stream>>>(
      WAp, xBp, bq, bk, bv, qT, kBp, vBp);
  attn_kernel<<<dim3(NB * NT / 4), dim3(128), 0, stream>>>(x, gamma, qT, kBp, vBp, out);
}
